// NormalizedDelinear_12249246728783
// MI455X (gfx1250) — compile-verified, matrix operand order checked by reading, not run on hardware
//
#include <hip/hip_runtime.h>

typedef __attribute__((ext_vector_type(2)))  float    v2f;
typedef __attribute__((ext_vector_type(8)))  float    v8f;
typedef __attribute__((ext_vector_type(8)))  __bf16   v8bf;
typedef __attribute__((ext_vector_type(16))) __bf16   v16bf;
typedef __attribute__((ext_vector_type(4)))  unsigned v4u;
typedef __attribute__((ext_vector_type(4)))  int      v4i;
typedef __attribute__((ext_vector_type(8)))  int      v8i;

#define EPSV   1e-5f
#define BLK    512
#define NROWS  131072      // 32768*2048/512
#define DDIM   2048
#define NSAMP  32768

// ---------------------------------------------------------------- utilities
__global__ void k_zero(float* __restrict__ p, int n) {
  int i = blockIdx.x * blockDim.x + threadIdx.x;
  if (i < n) p[i] = 0.f;
}

__global__ void k_colmean(const float* __restrict__ X, float* __restrict__ mean,
                          int rowsPerChunk) {
  int col = blockIdx.x * 256 + threadIdx.x;          // 0..511
  int r0  = blockIdx.y * rowsPerChunk;
  float s = 0.f;
  for (int r = 0; r < rowsPerChunk; ++r)
    s += X[(size_t)(r0 + r) * BLK + col];
  atomicAdd(mean + col, s);
}

__global__ void k_vscale(float* __restrict__ p, int n, float f) {
  int i = blockIdx.x * blockDim.x + threadIdx.x;
  if (i < n) p[i] *= f;
}

__global__ void k_cvt_bf16(const float* __restrict__ in, __bf16* __restrict__ o, int n) {
  int i = (blockIdx.x * 256 + threadIdx.x) * 4;
  if (i < n) {
    float4 v = *(const float4*)(in + i);
    o[i]     = (__bf16)v.x;
    o[i + 1] = (__bf16)v.y;
    o[i + 2] = (__bf16)v.z;
    o[i + 3] = (__bf16)v.w;
  }
}

// ------------------------------------------------- generic f32 WMMA matmul
// D[m,n] = sum_k A[m*rsA + k*csA] * B[k*rsB + n*csB]
// grid: (Mtiles64, Ntiles64, Ksplit); block: 256 (8 waves)
// mode: 0 = store, 1 = store(1.5*I - 0.5*acc), 2 = atomicAdd (split-K)
__global__ void k_matmul_f32(const float* __restrict__ A, const float* __restrict__ B,
                             float* __restrict__ Dm,
                             long rsA, long csA, long rsB, long csB,
                             int ldD, int kLen, int mode) {
  int tid  = threadIdx.x;
  int w    = tid >> 5;
  int lane = tid & 31;
  int lo   = lane & 15;
  int hi   = lane >> 4;
  int wm   = w & 3;
  int wn   = w >> 2;

  long row  = (long)blockIdx.x * 64 + wm * 16 + lo;
  long col0 = (long)blockIdx.y * 64 + wn * 32 + lo;
  long k0   = (long)blockIdx.z * kLen;

  v8f c0 = {0.f,0.f,0.f,0.f,0.f,0.f,0.f,0.f};
  v8f c1 = c0;

  const float* Arow = A + row * rsA;
#pragma unroll 8
  for (int kk = 0; kk < kLen; kk += 4) {
    long ka = k0 + kk + 2 * hi;          // f32 A 16x4: lanes0-15 K{0,1}, lanes16-31 K{2,3}
    v2f a, b0, b1;
    a.x  = Arow[ka * csA];
    a.y  = Arow[(ka + 1) * csA];
    b0.x = B[ka * rsB + col0 * csB];
    b0.y = B[(ka + 1) * rsB + col0 * csB];
    b1.x = B[ka * rsB + (col0 + 16) * csB];
    b1.y = B[(ka + 1) * rsB + (col0 + 16) * csB];
    c0 = __builtin_amdgcn_wmma_f32_16x16x4_f32(false, a, false, b0, (short)0, c0, false, false);
    c1 = __builtin_amdgcn_wmma_f32_16x16x4_f32(false, a, false, b1, (short)0, c1, false, false);
  }

  long orow = (long)blockIdx.x * 64 + wm * 16 + 8 * hi;   // C/D: VGPR v -> M=v(+8 hi lanes)
  long cA   = (long)blockIdx.y * 64 + wn * 32 + lo;
  long cB   = cA + 16;
#pragma unroll
  for (int v = 0; v < 8; ++v) {
    long r = orow + v;
    if (mode == 2) {
      atomicAdd(&Dm[r * ldD + cA], c0[v]);
      atomicAdd(&Dm[r * ldD + cB], c1[v]);
    } else if (mode == 1) {
      Dm[r * ldD + cA] = ((r == cA) ? 1.5f : 0.f) - 0.5f * c0[v];
      Dm[r * ldD + cB] = ((r == cB) ? 1.5f : 0.f) - 0.5f * c1[v];
    } else {
      Dm[r * ldD + cA] = c0[v];
      Dm[r * ldD + cB] = c1[v];
    }
  }
}

// ------------------------------------------- covariance finalize + ||A||_F^2
__global__ void k_cov_finalize(const float* __restrict__ acc, const float* __restrict__ mean,
                               float* __restrict__ cov, float* __restrict__ sumsq) {
  int i = blockIdx.x * 256 + threadIdx.x;          // 512*512 elements
  int p = i >> 9, q = i & 511;
  float c = acc[i] * (1.f / (float)NROWS) - mean[p] * mean[q] + ((p == q) ? EPSV : 0.f);
  cov[i] = c;
  float s = c * c;
  for (int m = 16; m; m >>= 1) s += __shfl_xor(s, m, 32);
  if ((threadIdx.x & 31) == 0) atomicAdd(sumsq, s);
}

__global__ void k_init_YZ(const float* __restrict__ cov, const float* __restrict__ sumsq,
                          float* __restrict__ Y, float* __restrict__ Z) {
  int i = blockIdx.x * 256 + threadIdx.x;
  int p = i >> 9, q = i & 511;
  float inv = rsqrtf(*sumsq);                      // 1/normA
  Y[i] = cov[i] * inv;
  Z[i] = (p == q) ? 1.f : 0.f;
}

__global__ void k_final_scale(const float* __restrict__ Zf, const float* __restrict__ sumsq,
                              float* __restrict__ o) {
  int i = blockIdx.x * 256 + threadIdx.x;
  o[i] = Zf[i] * rsqrtf(sqrtf(*sumsq));            // Z / sqrt(normA) = Z * S^-1/4
}

// ------------------------------------------------------------- bias fold
__global__ void k_bias(const float* __restrict__ wf, const float* __restrict__ mean,
                       const float* __restrict__ bias, float* __restrict__ bout) {
  int row  = blockIdx.x * 8 + (threadIdx.x >> 5);
  int lane = threadIdx.x & 31;
  float s = 0.f;
  const float* wr = wf + (size_t)row * DDIM;
  for (int k = lane; k < DDIM; k += 32) s += wr[k] * mean[k & 511];
  for (int m = 16; m; m >>= 1) s += __shfl_xor(s, m, 32);
  if (lane == 0) bout[row] = bias[row] - s;
}

// ===================== main GEMM: out = x @ Wf^T + b =====================
// A tiles: f32 via GLOBAL_LOAD_ASYNC_TO_LDS (ASYNCcnt), converted on read.
// B tiles: bf16 via Tensor Data Mover (TENSORcnt) with LDS row padding.

#define KS      32                       // K-step (floats / bf16 elements)
#define LDT     36                       // padded A LDS row stride (floats)
#define TILEF   (128 * LDT)              // floats per A tile
#define ATILE_B (TILEF * 4)              // 18432 bytes
#define BROW_B  80                       // B row: 64B data + 16B TDM pad
#define BTILE_B (128 * BROW_B)           // 10240 bytes
#define B_OFF   (2 * ATILE_B)            // byte offset of B tiles in LDS
#define SHMEM_B (2 * ATILE_B + 2 * BTILE_B)   // 57344 bytes

__device__ inline void async_copy16(unsigned ldsByteOff, const float* gptr) {
  asm volatile("global_load_async_to_lds_b128 %0, %1, off"
               :: "v"(ldsByteOff), "v"(gptr)
               : "memory");
}
__device__ inline void wait_async0() {
  asm volatile("s_wait_asynccnt 0x0" ::: "memory");
}

// Tensor DMA: load a 2D tile (32 bf16 x 128 rows) from a row-major
// [2048 x 2048] bf16 tensor into LDS, padding each 64B row to 80B.
// D# layout per CDNA5 ISA ch.8 (group0 128b, group1 256b; groups2/3 zero).
__device__ inline void tdm_load_B(const __bf16* gsrc, unsigned ldsByteOff) {
  unsigned long long ga = (unsigned long long)(uintptr_t)gsrc;
  v4u g0;
  g0[0] = 1u;                                           // count=1, user mode
  g0[1] = ldsByteOff;                                   // lds_addr
  g0[2] = (unsigned)(ga & 0xFFFFFFFFu);                 // global_addr[31:0]
  g0[3] = (unsigned)((ga >> 32) & 0x01FFFFFFu)          // global_addr[56:32]
        | (2u << 30);                                   // type=2 ("image")
  v8i g1;
  g1[0] = (1 << 16)                                     // data_size = 2 bytes
        | (1 << 20)                                     // pad_enable
        | (3 << 22)                                     // pad_interval: 16 DW = 64B
        | (3 << 25);                                    // pad_amount: 4 DW = 16B
  g1[1] = (int)((unsigned)DDIM << 16);                  // tensor_dim0[15:0] @ [31:16]
  g1[2] = (int)((unsigned)DDIM << 16);                  // dim0[31:16]=0 | tensor_dim1[15:0]
  g1[3] = (int)(32u << 16);                             // dim1[31:16]=0 | tile_dim0=32
  g1[4] = 128;                                          // tile_dim1=128, tile_dim2=0
  g1[5] = DDIM;                                         // tensor_dim0_stride[31:0]
  g1[6] = 0;                                            // stride hi / dim1_stride
  g1[7] = 0;
  v4i gz4 = {0, 0, 0, 0};
  v8i gz8 = {0, 0, 0, 0, 0, 0, 0, 0};
  __builtin_amdgcn_tensor_load_to_lds(g0, g1, gz4, gz4, gz8, 0);
}

// convert 8+8 contiguous floats (at p and p+16) to a bf16 frag
__device__ inline v16bf cvt_frag(const float* __restrict__ p) {
  float4 q0 = *(const float4*)(p);
  float4 q1 = *(const float4*)(p + 4);
  float4 q2 = *(const float4*)(p + 16);
  float4 q3 = *(const float4*)(p + 20);
  v16bf r;
  r[0]  = (__bf16)q0.x; r[1]  = (__bf16)q0.y; r[2]  = (__bf16)q0.z; r[3]  = (__bf16)q0.w;
  r[4]  = (__bf16)q1.x; r[5]  = (__bf16)q1.y; r[6]  = (__bf16)q1.z; r[7]  = (__bf16)q1.w;
  r[8]  = (__bf16)q2.x; r[9]  = (__bf16)q2.y; r[10] = (__bf16)q2.z; r[11] = (__bf16)q2.w;
  r[12] = (__bf16)q3.x; r[13] = (__bf16)q3.y; r[14] = (__bf16)q3.z; r[15] = (__bf16)q3.w;
  return r;
}

// grid (256, 16), block 256 (8 waves). Block tile 128x128, wave tile 32x64.
__global__ void k_gemm_bf16(const float* __restrict__ X, const __bf16* __restrict__ WfBf,
                            const float* __restrict__ Bv, float* __restrict__ Out) {
  extern __shared__ char smemc[];
  float* smem = (float*)smemc;

  int tid  = threadIdx.x;
  int w    = tid >> 5;
  int lane = tid & 31;
  int lo   = lane & 15;
  int hi   = lane >> 4;
  int wm   = w & 3;          // 4 waves along M
  int wn   = w >> 2;         // 2 waves along N

  long rowBase = (long)blockIdx.x * 128;
  long colBase = (long)blockIdx.y * 128;

  // A-tile copy slots: 4 float4 quads per thread (128 rows x 8 quads)
  int q0r[4], q0k[4];
#pragma unroll
  for (int i = 0; i < 4; ++i) {
    int q = tid + i * 256;
    q0r[i] = q >> 3;
    q0k[i] = (q & 7) * 4;
  }

  auto issueA = [&](int buf, int k) {
#pragma unroll
    for (int i = 0; i < 4; ++i) {
      int r = q0r[i], kq = q0k[i];
      unsigned lofs = (unsigned)((buf * TILEF + r * LDT + kq) * 4);
      async_copy16(lofs, X + (rowBase + r) * DDIM + k + kq);
    }
  };

  v8f z = {0.f,0.f,0.f,0.f,0.f,0.f,0.f,0.f};
  v8f c[2][4];
#pragma unroll
  for (int mi = 0; mi < 2; ++mi)
#pragma unroll
    for (int ni = 0; ni < 4; ++ni) c[mi][ni] = z;

  const int NK = DDIM / KS;          // 64
  issueA(0, 0);
  if (tid < 32)                      // wave0 drives the TDM engine
    tdm_load_B(WfBf + colBase * DDIM, (unsigned)B_OFF);

  for (int ks = 0; ks < NK; ++ks) {
    int cur = ks & 1;
    wait_async0();                                // my A-tile chunks landed
    if (tid < 32) __builtin_amdgcn_s_wait_tensorcnt(0);   // B tile landed
    __syncthreads();
    if (ks + 1 < NK) {
      issueA(1 - cur, (ks + 1) * KS);
      if (tid < 32)
        tdm_load_B(WfBf + colBase * DDIM + (ks + 1) * KS,
                   (unsigned)(B_OFF + (1 - cur) * BTILE_B));
    }

    const float* At  = smem + cur * TILEF;
    const char*  BtC = smemc + B_OFF + cur * BTILE_B;
    int kb = 8 * hi;     // frag: lanes0-15 K0-7/16-23, lanes16-31 K8-15/24-31

    v16bf a[2], b[4];
#pragma unroll
    for (int mi = 0; mi < 2; ++mi)
      a[mi] = cvt_frag(At + (wm * 32 + mi * 16 + lo) * LDT + kb);
#pragma unroll
    for (int ni = 0; ni < 4; ++ni) {
      const char* rp = BtC + (wn * 64 + ni * 16 + lo) * BROW_B + 16 * hi;
      v8bf l0 = *(const v8bf*)rp;
      v8bf h0 = *(const v8bf*)(rp + 32);
      b[ni] = __builtin_shufflevector(l0, h0, 0,1,2,3,4,5,6,7,8,9,10,11,12,13,14,15);
    }
#pragma unroll
    for (int mi = 0; mi < 2; ++mi)
#pragma unroll
      for (int ni = 0; ni < 4; ++ni)
        c[mi][ni] = __builtin_amdgcn_wmma_f32_16x16x32_bf16(
            false, a[mi], false, b[ni], (short)0, c[mi][ni], false, false);

    __syncthreads();                 // reads done before buffers are refilled
  }

#pragma unroll
  for (int mi = 0; mi < 2; ++mi)
#pragma unroll
    for (int ni = 0; ni < 4; ++ni) {
      long colO = colBase + wn * 64 + ni * 16 + lo;
      float bv  = Bv[colO];
#pragma unroll
      for (int v = 0; v < 8; ++v) {
        long rowO = rowBase + wm * 32 + mi * 16 + v + 8 * hi;
        Out[rowO * DDIM + colO] = c[mi][ni][v] + bv;
      }
    }
}

// ---------------------------------------------------------------- launcher
extern "C" void kernel_launch(void* const* d_in, const int* in_sizes, int n_in,
                              void* d_out, int out_size, void* d_ws, size_t ws_size,
                              hipStream_t stream) {
  (void)in_sizes; (void)n_in; (void)out_size; (void)ws_size;
  const float* x      = (const float*)d_in[0];   // [32768, 2048]
  const float* weight = (const float*)d_in[1];   // [2048, 2048]
  const float* bias   = (const float*)d_in[2];   // [2048]
  float* out = (float*)d_out;                    // [32768, 2048]
  float* ws  = (float*)d_ws;

  const int COVN = 512 * 512;                    // 262144
  float* mean   = ws;                            // 512
  float* sumsq  = ws + 512;                      // 1
  float* covAcc = ws + 1024;                     // 262144
  float* cov    = covAcc + COVN;
  float* Y0     = cov + COVN;
  float* Y1     = Y0 + COVN;
  float* Z0     = Y1 + COVN;
  float* Z1     = Z0 + COVN;
  float* Tm     = Z1 + COVN;
  float* cIs    = Tm + COVN;
  float* bf     = cIs + COVN;                    // 2048
  float* wf     = bf + 2048;                     // 2048*2048 f32
  __bf16* wfBf  = (__bf16*)(wf + DDIM * DDIM);   // 2048*2048 bf16

  // 1) zero mean/sumsq/covAcc
  k_zero<<<(1024 + COVN + 255) / 256, 256, 0, stream>>>(ws, 1024 + COVN);

  // 2) column means of X = x.reshape(-1, 512)
  k_colmean<<<dim3(2, 128), 256, 0, stream>>>(x, mean, NROWS / 128);
  k_vscale<<<2, 256, 0, stream>>>(mean, 512, 1.f / (float)NROWS);

  // 3) covAcc = X^T X (split-K=16, atomic).  A = X^T: rsA=1, csA=512.  B = X.
  k_matmul_f32<<<dim3(8, 8, 16), 256, 0, stream>>>(x, x, covAcc,
      1L, (long)BLK, (long)BLK, 1L, BLK, NROWS / 16, /*mode=*/2);

  // 4) cov = covAcc/n - m m^T + eps*I ; sumsq = ||cov||_F^2
  k_cov_finalize<<<COVN / 256, 256, 0, stream>>>(covAcc, mean, cov, sumsq);

  // 5) Y = cov/normA, Z = I
  k_init_YZ<<<COVN / 256, 256, 0, stream>>>(cov, sumsq, Y0, Z0);

  // 6) 5 Denman-Beavers iterations (f32 WMMA 512^3 matmuls)
  float* Yc = Y0; float* Yn = Y1; float* Zc = Z0; float* Zn = Z1;
  for (int it = 0; it < 5; ++it) {
    k_matmul_f32<<<dim3(8, 8, 1), 256, 0, stream>>>(Zc, Yc, Tm,
        (long)BLK, 1L, (long)BLK, 1L, BLK, BLK, /*mode=*/1);
    k_matmul_f32<<<dim3(8, 8, 1), 256, 0, stream>>>(Yc, Tm, Yn,
        (long)BLK, 1L, (long)BLK, 1L, BLK, BLK, /*mode=*/0);
    k_matmul_f32<<<dim3(8, 8, 1), 256, 0, stream>>>(Tm, Zc, Zn,
        (long)BLK, 1L, (long)BLK, 1L, BLK, BLK, /*mode=*/0);
    float* t;
    t = Yc; Yc = Yn; Yn = t;
    t = Zc; Zc = Zn; Zn = t;
  }

  // 7) cov_isqrt = Zc * normA^-1/2
  k_final_scale<<<COVN / 256, 256, 0, stream>>>(Zc, sumsq, cIs);

  // 8) wf = weight.reshape(8192,512) @ cov_isqrt
  k_matmul_f32<<<dim3(128, 8, 1), 256, 0, stream>>>(weight, cIs, wf,
      (long)BLK, 1L, (long)BLK, 1L, BLK, BLK, /*mode=*/0);

  // 9) b = bias - sum_k wf[o,k]*mean[k%512]
  k_bias<<<256, 256, 0, stream>>>(wf, mean, bias, bf);

  // 9b) wfBf = bf16(wf) for the main GEMM's B operand
  k_cvt_bf16<<<(DDIM * DDIM / 4 + 255) / 256, 256, 0, stream>>>(wf, wfBf, DDIM * DDIM);

  // 10) out = x @ wf^T + b  (bf16 WMMA; A via async-to-LDS, B via TDM)
  k_gemm_bf16<<<dim3(NSAMP / 128, DDIM / 128), 256, SHMEM_B, stream>>>(x, wfBf, bf, out);
}
